// nEq_NLMP2_18013092840061
// MI455X (gfx1250) — matrix operand order repaired and checked by reading, not compiled, not run on hardware
//
#include <hip/hip_runtime.h>

#define NNODES 50000
#define NEDGES 800000
#define DIM 64
#define EB 10
#define K1 224      // pad(3*DIM+EB = 202) up to multiple of 32
#define K2 128      // 2*DIM
#define FCH 256
#define MT 64       // edges per block (4 x 16-row WMMA tiles)

typedef __attribute__((ext_vector_type(16))) _Float16 v16h;
typedef __attribute__((ext_vector_type(8)))  _Float16 v8h;
typedef __attribute__((ext_vector_type(8)))  float    v8f;

__device__ __forceinline__ v16h frag_from_2x8(v8h lo, v8h hi) {
  v16h r;
  #pragma unroll
  for (int i = 0; i < 8; ++i) { r[i] = lo[i]; r[i + 8] = hi[i]; }
  return r;
}

// A fragment (16x32 f16, MxK) from row-major tile with `stride` halves per row.
// lane L: M = L&15 ; halves 0..7 -> K = k0 + (L>>4)*8 + j ; halves 8..15 -> +16
__device__ __forceinline__ v16h load_A(const _Float16* base, int stride, int k0, int lane) {
  const _Float16* p = base + (lane & 15) * stride + k0 + ((lane >> 4) << 3);
  return frag_from_2x8(*(const v8h*)p, *(const v8h*)(p + 16));
}

// B fragment (32x16, KxN) from K-major (transposed) weights wt[N][ld].
// lane L: N = n0 + (L&15) ; halves 0..7 -> K = k0 + (L>>4)*8 + j ; halves 8..15 -> +16
__device__ __forceinline__ v16h load_B(const _Float16* wt, int ld, int n0, int k0, int lane) {
  const _Float16* p = wt + (size_t)(n0 + (lane & 15)) * ld + k0 + ((lane >> 4) << 3);
  return frag_from_2x8(*(const v8h*)p, *(const v8h*)(p + 16));
}

__device__ __forceinline__ v8f wmma16(v16h a, v16h b, v8f c) {
  return __builtin_amdgcn_wmma_f32_16x16x32_f16(false, a, false, b, (short)0, c,
                                                false, false);
}

// -------------------------------------------------------------------------
// Workspace prep kernels
// -------------------------------------------------------------------------
__global__ void zero_f32(float* p, int n) {
  int i = blockIdx.x * blockDim.x + threadIdx.x;
  if (i < n) p[i] = 0.0f;
}

// src is K x N row-major f32; dst is N x Kpad row-major f16 (K-major per column,
// zero padded K..Kpad-1).
__global__ void convert_transpose(const float* __restrict__ src,
                                  _Float16* __restrict__ dst,
                                  int K, int N, int Kpad) {
  int idx = blockIdx.x * blockDim.x + threadIdx.x;
  if (idx >= N * Kpad) return;
  int n = idx / Kpad, k = idx % Kpad;
  dst[idx] = (k < K) ? (_Float16)src[(size_t)k * N + n] : (_Float16)0.0f;
}

// -------------------------------------------------------------------------
// Edge MLP: he_new = he + relu([he|hn[src]|hn[dst]|emb] @ W1a + b1a) @ W1b + b1b
// plus scatter-atomic of he_new*norm into node_ftr[dst]
// 64 edges per block: each B fragment reused across 4 A tiles.
// -------------------------------------------------------------------------
__global__ __launch_bounds__(256) void edge_mlp_kernel(
    const float* __restrict__ hn, const float* __restrict__ he,
    const float* __restrict__ emb, const float* __restrict__ norm,
    const int* __restrict__ src, const int* __restrict__ dst,
    const _Float16* __restrict__ W1aT, const float* __restrict__ b1a,
    const _Float16* __restrict__ W1bT, const float* __restrict__ b1b,
    float* __restrict__ he_out, float* __restrict__ node_ftr) {
  __shared__ _Float16 xs[MT * K1];     // 28 KB staged input tile (f16)
  __shared__ _Float16 hs[MT * FCH];    // 32 KB hidden activations (f16)
  __shared__ int      s_dst[MT];
  __shared__ float    s_norm[MT];

  const int tid = threadIdx.x;
  const int lane = tid & 31;
  const int wave = tid >> 5;
  const int e0 = blockIdx.x * MT;

  // Stage x = [he | hn[src] | hn[dst] | emb | 0...] as f16 (gathers hit L2)
  for (int idx = tid; idx < MT * K1; idx += 256) {
    int m = idx / K1, k = idx - m * K1;
    int e = e0 + m;
    float v;
    if (k < DIM)               v = he[(size_t)e * DIM + k];
    else if (k < 2 * DIM)      v = hn[(size_t)src[e] * DIM + (k - DIM)];
    else if (k < 3 * DIM)      v = hn[(size_t)dst[e] * DIM + (k - 2 * DIM)];
    else if (k < 3 * DIM + EB) v = emb[(size_t)e * EB + (k - 3 * DIM)];
    else                       v = 0.0f;
    xs[idx] = (_Float16)v;
  }
  if (tid < MT) { s_dst[tid] = dst[e0 + tid]; s_norm[tid] = norm[e0 + tid]; }
  __syncthreads();

  // GEMM1: [64 x 224] @ [224 x 256]; wave w owns hidden columns [32w, 32w+32)
  // for all four 16-row M tiles (B fragments reused 4x).
  {
    v8f acc[4][2];
    #pragma unroll
    for (int mt = 0; mt < 4; ++mt) { acc[mt][0] = (v8f){}; acc[mt][1] = (v8f){}; }
    const int n0 = wave * 32;
    #pragma unroll
    for (int kc = 0; kc < K1; kc += 32) {
      v16h b0 = load_B(W1aT, K1, n0,      kc, lane);
      v16h b1 = load_B(W1aT, K1, n0 + 16, kc, lane);
      #pragma unroll
      for (int mt = 0; mt < 4; ++mt) {
        v16h a = load_A(xs + mt * 16 * K1, K1, kc, lane);
        acc[mt][0] = wmma16(a, b0, acc[mt][0]);
        acc[mt][1] = wmma16(a, b1, acc[mt][1]);
      }
    }
    const int na = n0 + (lane & 15), nb = na + 16, hi = lane >> 4;
    const float ba = b1a[na], bb = b1a[nb];
    #pragma unroll
    for (int mt = 0; mt < 4; ++mt) {
      #pragma unroll
      for (int r = 0; r < 8; ++r) {
        int m = mt * 16 + r + hi * 8;
        float va = acc[mt][0][r] + ba; va = va > 0.0f ? va : 0.0f;
        float vb = acc[mt][1][r] + bb; vb = vb > 0.0f ? vb : 0.0f;
        hs[m * FCH + na] = (_Float16)va;
        hs[m * FCH + nb] = (_Float16)vb;
      }
    }
  }
  __syncthreads();

  // GEMM2: [64 x 256] @ [256 x 64] -> 4x4 output tiles; wave w does
  // N-tile (w&3) for M-tiles {2*(w>>2), 2*(w>>2)+1}.
  {
    const int nt = wave & 3, mtb = (wave >> 2) * 2;
    const int n0 = nt * 16;
    v8f c0 = {}, c1 = {};
    #pragma unroll
    for (int kc = 0; kc < FCH; kc += 32) {
      v16h b = load_B(W1bT, FCH, n0, kc, lane);
      c0 = wmma16(load_A(hs + (size_t)(mtb)     * 16 * FCH, FCH, kc, lane), b, c0);
      c1 = wmma16(load_A(hs + (size_t)(mtb + 1) * 16 * FCH, FCH, kc, lane), b, c1);
    }
    const int n = n0 + (lane & 15), hi = lane >> 4;
    const float bias = b1b[n];
    #pragma unroll
    for (int t = 0; t < 2; ++t) {
      v8f c = t ? c1 : c0;
      #pragma unroll
      for (int r = 0; r < 8; ++r) {
        int m = (mtb + t) * 16 + r + hi * 8;
        int e = e0 + m;
        float v = he[(size_t)e * DIM + n] + c[r] + bias;
        he_out[(size_t)e * DIM + n] = v;
        atomicAdd(&node_ftr[(size_t)s_dst[m] * DIM + n], v * s_norm[m]);
      }
    }
  }
}

// -------------------------------------------------------------------------
// Node MLP: hn_new = hn + relu([hn|node_ftr] @ W2a + b2a) @ W2b + b2b
// -------------------------------------------------------------------------
__global__ __launch_bounds__(256) void node_mlp_kernel(
    const float* __restrict__ hn, const float* __restrict__ node_ftr,
    const _Float16* __restrict__ W2aT, const float* __restrict__ b2a,
    const _Float16* __restrict__ W2bT, const float* __restrict__ b2b,
    float* __restrict__ hn_out) {
  __shared__ _Float16 ys[16 * K2];
  __shared__ _Float16 hs[16 * FCH];

  const int tid = threadIdx.x;
  const int lane = tid & 31;
  const int wave = tid >> 5;
  const int v0 = blockIdx.x * 16;

  for (int idx = tid; idx < 16 * K2; idx += 256) {
    int m = idx / K2, k = idx - m * K2;
    int v = v0 + m;
    float val = (k < DIM) ? hn[(size_t)v * DIM + k]
                          : node_ftr[(size_t)v * DIM + (k - DIM)];
    ys[idx] = (_Float16)val;
  }
  __syncthreads();

  {
    v8f c0 = {}, c1 = {};
    const int n0 = wave * 32;
    #pragma unroll
    for (int kc = 0; kc < K2; kc += 32) {
      v16h a = load_A(ys, K2, kc, lane);
      c0 = wmma16(a, load_B(W2aT, K2, n0,      kc, lane), c0);
      c1 = wmma16(a, load_B(W2aT, K2, n0 + 16, kc, lane), c1);
    }
    const int na = n0 + (lane & 15), nb = na + 16, hi = lane >> 4;
    const float ba = b2a[na], bb = b2a[nb];
    #pragma unroll
    for (int r = 0; r < 8; ++r) {
      int m = r + hi * 8;
      float va = c0[r] + ba; va = va > 0.0f ? va : 0.0f;
      float vb = c1[r] + bb; vb = vb > 0.0f ? vb : 0.0f;
      hs[m * FCH + na] = (_Float16)va;
      hs[m * FCH + nb] = (_Float16)vb;
    }
  }
  __syncthreads();

  if (wave < 4) {
    const int n0 = wave * 16;
    v8f c = {};
    #pragma unroll
    for (int kc = 0; kc < FCH; kc += 32)
      c = wmma16(load_A(hs, FCH, kc, lane), load_B(W2bT, FCH, n0, kc, lane), c);
    const int n = n0 + (lane & 15), hi = lane >> 4;
    const float bias = b2b[n];
    #pragma unroll
    for (int r = 0; r < 8; ++r) {
      int m = r + hi * 8, v = v0 + m;
      hn_out[(size_t)v * DIM + n] = hn[(size_t)v * DIM + n] + c[r] + bias;
    }
  }
}

// -------------------------------------------------------------------------
extern "C" void kernel_launch(void* const* d_in, const int* in_sizes, int n_in,
                              void* d_out, int out_size, void* d_ws, size_t ws_size,
                              hipStream_t stream) {
  (void)in_sizes; (void)n_in; (void)out_size; (void)ws_size;
  const float* hn   = (const float*)d_in[0];
  const float* he   = (const float*)d_in[1];
  const float* emb  = (const float*)d_in[2];
  const float* norm = (const float*)d_in[3];
  const int*   eidx = (const int*)d_in[4];
  const float* W1a  = (const float*)d_in[5];
  const float* b1a  = (const float*)d_in[6];
  const float* W1b  = (const float*)d_in[7];
  const float* b1b  = (const float*)d_in[8];
  const float* W2a  = (const float*)d_in[9];
  const float* b2a  = (const float*)d_in[10];
  const float* W2b  = (const float*)d_in[11];
  const float* b2b  = (const float*)d_in[12];

  const int* src = eidx;            // edge_index[0]
  const int* dst = eidx + NEDGES;   // edge_index[1]

  float* hn_out = (float*)d_out;                         // [NNODES, DIM]
  float* he_out = hn_out + (size_t)NNODES * DIM;         // [NEDGES, DIM]

  // workspace layout (16B-aligned segments)
  float*    node_ftr = (float*)d_ws;                             // NNODES*DIM f32
  _Float16* W1aT = (_Float16*)(node_ftr + (size_t)NNODES * DIM); // [256][224]
  _Float16* W1bT = W1aT + (size_t)FCH * K1;                      // [64][256]
  _Float16* W2aT = W1bT + (size_t)DIM * FCH;                     // [256][128]
  _Float16* W2bT = W2aT + (size_t)FCH * K2;                      // [64][256]

  zero_f32<<<(NNODES * DIM + 255) / 256, 256, 0, stream>>>(node_ftr, NNODES * DIM);
  convert_transpose<<<(FCH * K1 + 255) / 256, 256, 0, stream>>>(W1a, W1aT, 3 * DIM + EB, FCH, K1);
  convert_transpose<<<(DIM * FCH + 255) / 256, 256, 0, stream>>>(W1b, W1bT, FCH, DIM, FCH);
  convert_transpose<<<(FCH * K2 + 255) / 256, 256, 0, stream>>>(W2a, W2aT, K2, FCH, K2);
  convert_transpose<<<(DIM * FCH + 255) / 256, 256, 0, stream>>>(W2b, W2bT, FCH, DIM, FCH);

  edge_mlp_kernel<<<NEDGES / MT, 256, 0, stream>>>(hn, he, emb, norm, src, dst,
                                                   W1aT, b1a, W1bT, b1b,
                                                   he_out, node_ftr);
  node_mlp_kernel<<<NNODES / 16, 256, 0, stream>>>(hn, node_ftr,
                                                   W2aT, b2a, W2bT, b2b, hn_out);
}
